// PointConvNet_69458211111248
// MI455X (gfx1250) — compile-verified, matrix operand order checked by reading, not run on hardware
//
#include <hip/hip_runtime.h>
#include <hip/hip_bf16.h>

typedef __attribute__((ext_vector_type(16))) __bf16 v16bf;
typedef __attribute__((ext_vector_type(8)))  __bf16 v8bf;
typedef __attribute__((ext_vector_type(8)))  float  v8f;

#define N_NODES 50000
#define N_EDGES 800000
#define IN_CH   32
#define HID     64
#define OUT_CH  64
#define N_TILES (N_EDGES / 16)          // 50000 exact (no partial tiles)
#define OUT_ELEMS ((long)N_NODES * OUT_CH)
#define WFRAG_ELEMS (16 * 32 * 16)      // 16 fragments x 32 lanes x 16 bf16

// ---------------------------------------------------------------------------
// Fragment element e (0..15) + lane half -> local K index, per CDNA5 ISA
// 16-bit A/B layout (§7.12.2):
//   e in [0,8)  : k = koff + e
//   e in [8,16) : k = koff + 16 + (e-8)        (koff = lane<16 ? 0 : 8)
// ---------------------------------------------------------------------------

// Pre-swizzle W1 (35x64, K zero-padded to 64) and W2 (64x64) into bf16
// B-fragments. Fragment f = layer*8 + kk*4 + nt; storage: wf[f][lane][e].
__global__ void prep_weights(const float* __restrict__ W1,
                             const float* __restrict__ W2,
                             __bf16* __restrict__ wf) {
  int t = blockIdx.x * blockDim.x + threadIdx.x;       // 0..8191
  if (t >= WFRAG_ELEMS) return;
  int f    = t >> 9;
  int lane = (t >> 4) & 31;
  int e    = t & 15;
  int layer = f >> 3;
  int kk    = (f >> 2) & 1;
  int nt    = f & 3;
  int n     = nt * 16 + (lane & 15);
  int koff  = (lane < 16) ? 0 : 8;
  int klocal = (e < 8) ? (koff + e) : (koff + 16 + (e - 8));
  int k = kk * 32 + klocal;
  float val;
  if (layer == 0) val = (k < IN_CH + 3) ? W1[k * HID + n] : 0.0f;
  else            val = W2[k * OUT_CH + n];
  wf[t] = (__bf16)val;
}

// Init out region to -inf (segment-max identity) and emit pos3 output.
__global__ void init_out(float* __restrict__ out, const float* __restrict__ pos) {
  long stride = (long)gridDim.x * blockDim.x;
  long i = (long)blockIdx.x * blockDim.x + threadIdx.x;
  for (long j = i; j < OUT_ELEMS; j += stride)
    out[j] = -__builtin_inff();
  float* p3 = out + OUT_ELEMS;
  for (long n = i; n < N_NODES; n += stride) {
    float z   = pos[n * 2 + 0];
    float phi = pos[n * 2 + 1];
    p3[n * 3 + 0] = __cosf(phi);
    p3[n * 3 + 1] = __sinf(phi);
    p3[n * 3 + 2] = z;
  }
}

__device__ __forceinline__ v8f wmma_bf16(v16bf a, v16bf b, v8f c) {
  return __builtin_amdgcn_wmma_f32_16x16x32_bf16(
      /*neg_a=*/false, a, /*neg_b=*/false, b,
      /*c_mod=*/(short)0, c, /*reuse_a=*/false, /*reuse_b=*/false);
}

// Load one 32-byte bf16 fragment from LDS (lowers to 2x ds_load_b128).
__device__ __forceinline__ v16bf load_frag(const __bf16* p) {
  v8bf lo = *(const v8bf*)(p);
  v8bf hi = *(const v8bf*)(p + 8);
  v16bf r;
  for (int i = 0; i < 8; ++i) { r[i] = lo[i]; r[8 + i] = hi[i]; }
  return r;
}

// One wave32 = one tile of 16 edges; full 16x64 MLP via 16 bf16 WMMAs,
// scatter via native f32 atomic-max. Weights live in LDS, not VGPRs,
// to maximize occupancy for this latency-bound (atomics+gathers) kernel.
__global__ void __launch_bounds__(256)
edge_mlp(const float* __restrict__ x,
         const float* __restrict__ pos,
         const int*   __restrict__ ei,
         const __bf16* __restrict__ wf,
         const float* __restrict__ b1,
         const float* __restrict__ b2,
         float* __restrict__ out) {
  __shared__ __bf16 wfLds[WFRAG_ELEMS];      // 16 KB swizzled weight fragments
  __shared__ __bf16 hT[8][16][HID];          // 2 KB/wave transpose staging

  const int lane = threadIdx.x & 31;
  const int wave = threadIdx.x >> 5;
  const int totalWaves = (gridDim.x * blockDim.x) >> 5;
  const int gwave = (blockIdx.x * blockDim.x + threadIdx.x) >> 5;

  // Cooperative weight stage-in (uniform control flow), then one barrier.
  for (int t = threadIdx.x; t < WFRAG_ELEMS / 8; t += blockDim.x)
    ((v8bf*)wfLds)[t] = ((const v8bf*)wf)[t];
  __syncthreads();

  const int m    = lane & 15;                // tile row this lane owns
  const int koff = (lane < 16) ? 0 : 8;
  const int cm   = (lane < 16) ? 0 : 8;      // C-layout row base

  const __bf16* w1p = wfLds + (long)lane * 16;
  const __bf16* w2p = wfLds + 8 * 512 + (long)lane * 16;

  float b1v[4], b2v[4];
  for (int nt = 0; nt < 4; ++nt) {
    b1v[nt] = b1[nt * 16 + m];
    b2v[nt] = b2[nt * 16 + m];
  }

  for (int tile = gwave; tile < N_TILES; tile += totalWaves) {
    const int e   = tile * 16 + m;
    const int src = ei[e];
    const int dst = ei[N_EDGES + e];

    // Prefetch next tile's edge indices while we chew on this one.
    int eNext = e + totalWaves * 16;
    if (eNext < N_EDGES) {
      __builtin_prefetch(ei + eNext, 0, 0);
      __builtin_prefetch(ei + N_EDGES + eNext, 0, 0);
    }

    // ---- A fragment, K-step 0: x[src] features 0..31 (each row read once) --
    const float4* xp = (const float4*)(x + (long)src * IN_CH);
    int fb = koff >> 2;
    float4 f0 = xp[fb], f1 = xp[fb + 1], f2 = xp[fb + 4], f3 = xp[fb + 5];
    v16bf a1;
    a1[0]  = (__bf16)f0.x; a1[1]  = (__bf16)f0.y; a1[2]  = (__bf16)f0.z; a1[3]  = (__bf16)f0.w;
    a1[4]  = (__bf16)f1.x; a1[5]  = (__bf16)f1.y; a1[6]  = (__bf16)f1.z; a1[7]  = (__bf16)f1.w;
    a1[8]  = (__bf16)f2.x; a1[9]  = (__bf16)f2.y; a1[10] = (__bf16)f2.z; a1[11] = (__bf16)f2.w;
    a1[12] = (__bf16)f3.x; a1[13] = (__bf16)f3.y; a1[14] = (__bf16)f3.z; a1[15] = (__bf16)f3.w;

    // ---- A fragment, K-step 1: geometric features at k=32..34, rest 0 ----
    float2 ps = *(const float2*)(pos + (long)src * 2);
    float2 pd = *(const float2*)(pos + (long)dst * 2);
    float relx = __cosf(ps.y) - __cosf(pd.y);
    float rely = __sinf(ps.y) - __sinf(pd.y);
    float relz = ps.x - pd.x;
    v16bf a2;
    for (int i = 0; i < 16; ++i) a2[i] = (__bf16)0.0f;
    if (lane < 16) {                         // k=32,33,34 -> low-half lanes, e=0..2
      a2[0] = (__bf16)relx;
      a2[1] = (__bf16)rely;
      a2[2] = (__bf16)relz;
    }

    // ---- Layer 1: h = relu(msg_in @ W1 + b1) -> LDS for layout transpose --
    for (int nt = 0; nt < 4; ++nt) {
      v8f c;
      for (int v = 0; v < 8; ++v) c[v] = b1v[nt];
      c = wmma_bf16(a1, load_frag(w1p + (0 * 4 + nt) * 512), c);
      c = wmma_bf16(a2, load_frag(w1p + (1 * 4 + nt) * 512), c);
      int ncol = nt * 16 + m;
      for (int v = 0; v < 8; ++v) {
        float cv = c[v];
        float hv = cv > 0.0f ? cv : 0.0f;    // ReLU (finite data; no NaN path)
        hT[wave][cm + v][ncol] = (__bf16)hv;
      }
    }
    asm volatile("s_wait_dscnt 0" ::: "memory");   // wave-local LDS RAW fence

    // ---- Re-load h as layer-2 A fragments (transpose via LDS) ----
    const __bf16* row = &hT[wave][m][0];
    v16bf ha[2];
    for (int kk = 0; kk < 2; ++kk) {
      v8bf lo = *(const v8bf*)(row + kk * 32 + koff);        // ds_load_b128
      v8bf hi = *(const v8bf*)(row + kk * 32 + koff + 16);   // ds_load_b128
      for (int i = 0; i < 8; ++i) { ha[kk][i] = lo[i]; ha[kk][8 + i] = hi[i]; }
    }

    // ---- Layer 2 + scatter segment-max ----
    for (int nt = 0; nt < 4; ++nt) {
      v8f c;
      for (int v = 0; v < 8; ++v) c[v] = b2v[nt];
      c = wmma_bf16(ha[0], load_frag(w2p + (0 * 4 + nt) * 512), c);
      c = wmma_bf16(ha[1], load_frag(w2p + (1 * 4 + nt) * 512), c);
      int ncol = nt * 16 + m;
      for (int v = 0; v < 8; ++v) {
        int dstm = __shfl(dst, cm + v, 32);  // dst node of row cm+v
        unsigned off = ((unsigned)dstm * OUT_CH + (unsigned)ncol) * 4u;
        float val = c[v];
        // no-return f32 atomic max (STOREcnt-tracked), saddr addressing
        asm volatile("global_atomic_max_num_f32 %0, %1, %2"
                     :: "v"(off), "v"(val), "s"(out) : "memory");
      }
    }
  }
}

// Nodes with no incoming edges remain -inf -> 0 (matches isfinite() select).
__global__ void finalize(float* __restrict__ out) {
  long stride = (long)gridDim.x * blockDim.x;
  for (long j = (long)blockIdx.x * blockDim.x + threadIdx.x; j < OUT_ELEMS; j += stride) {
    float v = out[j];
    out[j] = __builtin_isfinite(v) ? v : 0.0f;
  }
}

extern "C" void kernel_launch(void* const* d_in, const int* in_sizes, int n_in,
                              void* d_out, int out_size, void* d_ws, size_t ws_size,
                              hipStream_t stream) {
  const float* x   = (const float*)d_in[0];
  const float* pos = (const float*)d_in[1];
  const int*   ei  = (const int*)d_in[2];
  const float* W1  = (const float*)d_in[3];
  const float* b1  = (const float*)d_in[4];
  const float* W2  = (const float*)d_in[5];
  const float* b2  = (const float*)d_in[6];
  float* out = (float*)d_out;
  __bf16* wf = (__bf16*)d_ws;                // 16 KB of fragment storage

  prep_weights<<<32, 256, 0, stream>>>(W1, W2, wf);
  init_out<<<1024, 256, 0, stream>>>(out, pos);
  edge_mlp<<<1024, 256, 0, stream>>>(x, pos, ei, wf, b1, b2, out);
  finalize<<<1024, 256, 0, stream>>>(out);
}
